// GPT_34875134443647
// MI455X (gfx1250) — matrix operand order repaired and checked by reading, not compiled, not run on hardware
//
#include <hip/hip_runtime.h>

// GPT forward (6-layer, E=384, T=256, B=128, H=6, FF=1536, V=65) for gfx1250.
// All GEMMs run on v_wmma_f32_16x16x32_f16 (wave32, 16x16 tile per wave-op).
// GEMM wave tile = 16x128 (8 accumulators): per 32-K chunk -> 1 A-fragment
// load + 8 B-fragment loads + 8 back-to-back independent WMMAs (keeps the
// XDL pipe fed while VMEM co-executes). Weights are converted fp32 -> f16 and
// transposed to K-major once per launch so B-fragments are contiguous 16B
// loads. Activations f16 on the matrix path, f32 residual/LN/softmax/loss.

#define V_   65
#define E_   384
#define TC_  256
#define H_   6
#define HS_  64
#define L_   6
#define FF_  1536
#define B_   128
#define MTOK (B_ * TC_)        // 32768 rows
#define EPS_ 1e-5f
#define VPAD 128               // padded LM-head columns (multiple of 128)
#define NSUB 8                 // 8 x 16 = 128 N-columns per wave

typedef __attribute__((ext_vector_type(16))) _Float16 v16h;
typedef __attribute__((ext_vector_type(8)))  _Float16 v8h;
typedef __attribute__((ext_vector_type(8)))  float    v8f;

__device__ __forceinline__ v8f wmma_f16(v16h a, v16h b, v8f c) {
  // 8 args: (neg_a, A, neg_b, B, c_mod, C, reuse_a, reuse_b)
  return __builtin_amdgcn_wmma_f32_16x16x32_f16(false, a, false, b, (short)0, c,
                                                false, false);
}

// ---------------------------------------------------------------------------
// Weight convert + transpose: W[K,N] fp32 -> out[Npad][K] f16 (zero-pad n>=N)
// ---------------------------------------------------------------------------
__global__ void k_convT(const float* __restrict__ W, _Float16* __restrict__ out,
                        int K, int N, int Npad) {
  int id = blockIdx.x * 256 + threadIdx.x;
  if (id >= Npad * K) return;
  int n = id / K;
  int k = id - n * K;
  out[id] = (n < N) ? (_Float16)W[(size_t)k * N + n] : (_Float16)0.f;
}

// ---------------------------------------------------------------------------
// Embedding: x[row,:] = tok_emb[idx[row],:] + pos_emb[row % T,:]
// ---------------------------------------------------------------------------
__global__ void k_embed(const float* __restrict__ tok,
                        const float* __restrict__ pos,
                        const int* __restrict__ idx, float* __restrict__ x) {
  int row = blockIdx.x;
  int id = idx[row];
  int p = row % TC_;
  for (int e = threadIdx.x; e < E_; e += blockDim.x)
    x[(size_t)row * E_ + e] =
        tok[(size_t)id * E_ + e] + pos[(size_t)p * E_ + e];
}

// ---------------------------------------------------------------------------
// LayerNorm over E=384, output f16 for the WMMA path. One row per block.
// ---------------------------------------------------------------------------
__global__ void k_ln(const float* __restrict__ x, const float* __restrict__ s,
                     const float* __restrict__ b, _Float16* __restrict__ out) {
  __shared__ float r1[128], r2[128];
  int row = blockIdx.x, t = threadIdx.x;
  const float* xr = x + (size_t)row * E_;
  float sum = 0.f, sq = 0.f;
  for (int e = t; e < E_; e += 128) {
    float v = xr[e];
    sum += v;
    sq += v * v;
  }
  r1[t] = sum; r2[t] = sq;
  __syncthreads();
  for (int off = 64; off > 0; off >>= 1) {
    if (t < off) { r1[t] += r1[t + off]; r2[t] += r2[t + off]; }
    __syncthreads();
  }
  float mu = r1[0] * (1.f / E_);
  float var = r2[0] * (1.f / E_) - mu * mu;
  float inv = rsqrtf(var + EPS_);
  _Float16* o = out + (size_t)row * E_;
  for (int e = t; e < E_; e += 128)
    o[e] = (_Float16)((xr[e] - mu) * inv * s[e] + b[e]);
}

// ---------------------------------------------------------------------------
// WMMA GEMM: C[M,Nout] = A[M,K] @ Bt^T (+bias)(+resid)(relu?)
//   A   : f16 row-major, lda = K
//   Bt  : f16 K-major   [Npad][K]  (column n contiguous in K)
//   out : f32 and/or f16 (ldc = Nout), store guarded by n < Nout
// Block = 128 thr = 4 waves; wave tile 16(M) x 128(N) via 8 accumulators;
// block tile 64(M) x 128(N).  Grid = (M/64, Npad/128).
// ---------------------------------------------------------------------------
__global__ void k_gemm(const _Float16* __restrict__ A,
                       const _Float16* __restrict__ Bt,
                       const float* __restrict__ bias,
                       const float* __restrict__ resid,
                       float* __restrict__ out32, _Float16* __restrict__ out16,
                       int K, int Nout, int relu) {
  const int lane = threadIdx.x & 31;
  const int wave = threadIdx.x >> 5;
  const int half = lane >> 4;     // ISA: lanes 16-31 hold the other K sub-runs
  const int r = lane & 15;
  const int m0 = blockIdx.x * 64 + wave * 16;
  const int n0 = blockIdx.y * 128;

  const _Float16* arow = A + (size_t)(m0 + r) * K;    // row m0+r of A
  const _Float16* bcol0 = Bt + (size_t)(n0 + r) * K;  // col n0+r, K-contiguous

  v8f cacc[NSUB];
#pragma unroll
  for (int t = 0; t < NSUB; ++t) cacc[t] = (v8f){};

  for (int kc = 0; kc < K; kc += 32) {
    if (kc + 64 < K) __builtin_prefetch(arow + kc + 64, 0, 1);
    // A 16x32 fragment: runs [kc+8h, +8) and [kc+16+8h, +8)
    v8h alo = *(const v8h*)(arow + kc + 8 * half);
    v8h ahi = *(const v8h*)(arow + kc + 16 + 8 * half);
    v16h a;
#pragma unroll
    for (int i = 0; i < 8; ++i) { a[i] = alo[i]; a[8 + i] = ahi[i]; }
    // 8 independent B tiles / WMMAs sharing the A fragment
#pragma unroll
    for (int t = 0; t < NSUB; ++t) {
      const _Float16* bc = bcol0 + (size_t)(16 * t) * K + kc + 8 * half;
      v8h blo = *(const v8h*)bc;
      v8h bhi = *(const v8h*)(bc + 16);
      v16h bb;
#pragma unroll
      for (int i = 0; i < 8; ++i) { bb[i] = blo[i]; bb[8 + i] = bhi[i]; }
      cacc[t] = wmma_f16(a, bb, cacc[t]);
    }
  }

#pragma unroll
  for (int t = 0; t < NSUB; ++t) {
    int n = n0 + 16 * t + r;
    if (n < Nout) {
      float bv = bias ? bias[n] : 0.f;
#pragma unroll
      for (int i = 0; i < 8; ++i) {
        int m = m0 + i + 8 * half;  // C layout: VGPR i -> M=i (+8 hi lanes)
        float v = cacc[t][i] + bv;
        if (resid) v += resid[(size_t)m * Nout + n];
        if (relu) v = fmaxf(v, 0.f);
        if (out32) out32[(size_t)m * Nout + n] = v;
        if (out16) out16[(size_t)m * Nout + n] = (_Float16)v;
      }
    }
  }
}

// ---------------------------------------------------------------------------
// Causal attention, one wave per (batch, head, 16-query strip).
// scores = (Q K^T) * E^-0.5 via WMMA (K=64 -> 2 chunks), softmax in LDS,
// O = P @ V via WMMA over padded key chunks (A-fragments from LDS).
// q/k/v/o are [B,T,E] f16 with head h at columns h*64..h*64+63.
// ---------------------------------------------------------------------------
__global__ void k_attn(const _Float16* __restrict__ q,
                       const _Float16* __restrict__ k,
                       const _Float16* __restrict__ v,
                       _Float16* __restrict__ o) {
  __shared__ float sc[16][TC_];       // 16 KB
  __shared__ _Float16 pm[16][TC_];    // 8 KB
  const int lane = threadIdx.x & 31;
  const int half = lane >> 4;
  const int r = lane & 15;
  const int qt = blockIdx.x;          // query tile (16 rows)
  const int hh = blockIdx.y;
  const int bb_ = blockIdx.z;
  const float scale = 0.051031036307982884f;  // 384^-0.5 (NOTE: E, not HS)
  const size_t base = ((size_t)bb_ * TC_) * E_ + (size_t)hh * HS_;
  const _Float16* qp = q + base;
  const _Float16* kp = k + base;
  const _Float16* vp = v + base;

  // Q fragments for rows qt*16 .. +15, head-dim chunks [0,32) and [32,64)
  v16h aq0, aq1;
  {
    const _Float16* qr = qp + (size_t)(qt * 16 + r) * E_;
    v8h l0 = *(const v8h*)(qr + 8 * half);
    v8h h0 = *(const v8h*)(qr + 16 + 8 * half);
    v8h l1 = *(const v8h*)(qr + 32 + 8 * half);
    v8h h1 = *(const v8h*)(qr + 48 + 8 * half);
#pragma unroll
    for (int i = 0; i < 8; ++i) {
      aq0[i] = l0[i]; aq0[8 + i] = h0[i];
      aq1[i] = l1[i]; aq1[8 + i] = h1[i];
    }
  }

  // ---- scores for causal key tiles jt <= qt ----
  for (int jt = 0; jt <= qt; ++jt) {
    const _Float16* kr = kp + (size_t)(jt * 16 + r) * E_;  // key jt*16+r
    v8h b0 = *(const v8h*)(kr + 8 * half);
    v8h b1 = *(const v8h*)(kr + 16 + 8 * half);
    v8h b2 = *(const v8h*)(kr + 32 + 8 * half);
    v8h b3 = *(const v8h*)(kr + 48 + 8 * half);
    v16h bb0, bb1;
#pragma unroll
    for (int i = 0; i < 8; ++i) {
      bb0[i] = b0[i]; bb0[8 + i] = b1[i];
      bb1[i] = b2[i]; bb1[8 + i] = b3[i];
    }
    v8f c = {};
    c = wmma_f16(aq0, bb0, c);
    c = wmma_f16(aq1, bb1, c);
#pragma unroll
    for (int i = 0; i < 8; ++i) {
      int M = i + 8 * half;
      int key = jt * 16 + r;
      int qrow = qt * 16 + M;
      sc[M][jt * 16 + r] = (key <= qrow) ? c[i] * scale : -3.0e38f;
    }
  }
  __syncthreads();

  // ---- row softmax over valid keys, P -> f16 in LDS ----
  const int nkeys = (qt + 1) * 16;
  const int padded = (nkeys + 31) & ~31;
  if (lane < 16) {
    int row = lane;
    float mx = -3.0e38f;
    for (int kk = 0; kk < nkeys; ++kk) mx = fmaxf(mx, sc[row][kk]);
    float s = 0.f;
    for (int kk = 0; kk < nkeys; ++kk) {
      float e = expf(sc[row][kk] - mx);
      sc[row][kk] = e;
      s += e;
    }
    float inv = 1.f / s;
    for (int kk = 0; kk < nkeys; ++kk)
      pm[row][kk] = (_Float16)(sc[row][kk] * inv);
    for (int kk = nkeys; kk < padded; ++kk) pm[row][kk] = (_Float16)0.f;
  }
  __syncthreads();

  // ---- O = P @ V ----
  _Float16* op = o + base;
  for (int nt = 0; nt < 4; ++nt) {
    v8f c = {};
    for (int kc = 0; kc < padded; kc += 32) {
      v16h a, bb;
#pragma unroll
      for (int i = 0; i < 8; ++i) {
        a[i] = pm[r][kc + 8 * half + i];
        a[8 + i] = pm[r][kc + 16 + 8 * half + i];
      }
#pragma unroll
      for (int j = 0; j < 16; ++j)
        bb[j] = vp[(size_t)(kc + 8 * half + (j & 7) + 16 * (j >> 3)) * E_ + nt * 16 + r];
      c = wmma_f16(a, bb, c);
    }
#pragma unroll
    for (int i = 0; i < 8; ++i) {
      int M = i + 8 * half;
      op[(size_t)(qt * 16 + M) * E_ + nt * 16 + r] = (_Float16)c[i];
    }
  }
}

// ---------------------------------------------------------------------------
// Loss: mean over rows of -log_softmax(logits)[target]
// ---------------------------------------------------------------------------
__global__ void k_zero(float* p) {
  if (threadIdx.x == 0 && blockIdx.x == 0) *p = 0.f;
}

__global__ void k_loss(const float* __restrict__ logits,
                       const int* __restrict__ tgt, float* __restrict__ loss) {
  __shared__ float red[256];
  int i = blockIdx.x * 256 + threadIdx.x;
  float local = 0.f;
  if (i < MTOK) {
    const float* lr = logits + (size_t)i * V_;
    float mx = -3.0e38f;
    for (int j = 0; j < V_; ++j) mx = fmaxf(mx, lr[j]);
    float s = 0.f;
    for (int j = 0; j < V_; ++j) s += expf(lr[j] - mx);
    int t = tgt[i];
    float lp = lr[t] - mx - logf(s);
    local = -lp * (1.f / (float)MTOK);
  }
  red[threadIdx.x] = local;
  __syncthreads();
  for (int off = 128; off > 0; off >>= 1) {
    if (threadIdx.x < off) red[threadIdx.x] += red[threadIdx.x + off];
    __syncthreads();
  }
  if (threadIdx.x == 0) atomicAdd(loss, red[0]);
}

// ---------------------------------------------------------------------------
// Host driver
// Input order (setup_inputs dict, flattened):
//  0 tok_emb  1 pos_emb  2 Wq  3 Wk  4 Wv  5 Wo  6 bo  7 ln1_s  8 ln1_b
//  9 W1  10 b1  11 W2  12 b2  13 ln2_s  14 ln2_b  15 lnf_s  16 lnf_b
// 17 Wlm  18 blm  19 idx  20 targets
// d_out = [logits_flat (32768*65 f32) | loss (1 f32)]
// ---------------------------------------------------------------------------
extern "C" void kernel_launch(void* const* d_in, const int* in_sizes, int n_in,
                              void* d_out, int out_size, void* d_ws,
                              size_t ws_size, hipStream_t stream) {
  (void)in_sizes; (void)n_in; (void)out_size; (void)ws_size;
  const float* tok   = (const float*)d_in[0];
  const float* pos   = (const float*)d_in[1];
  const float* Wq    = (const float*)d_in[2];
  const float* Wk    = (const float*)d_in[3];
  const float* Wv    = (const float*)d_in[4];
  const float* Wo    = (const float*)d_in[5];
  const float* bo    = (const float*)d_in[6];
  const float* ln1s  = (const float*)d_in[7];
  const float* ln1b  = (const float*)d_in[8];
  const float* W1    = (const float*)d_in[9];
  const float* b1    = (const float*)d_in[10];
  const float* W2    = (const float*)d_in[11];
  const float* b2    = (const float*)d_in[12];
  const float* ln2s  = (const float*)d_in[13];
  const float* ln2b  = (const float*)d_in[14];
  const float* lnfs  = (const float*)d_in[15];
  const float* lnfb  = (const float*)d_in[16];
  const float* Wlm   = (const float*)d_in[17];
  const float* blm   = (const float*)d_in[18];
  const int*   idx   = (const int*)d_in[19];
  const int*   tgt   = (const int*)d_in[20];

  float* logits = (float*)d_out;
  float* loss = logits + (size_t)MTOK * V_;

  // Workspace carve-out (aligned to 256B)
  char* p = (char*)d_ws;
  auto alloc = [&](size_t bytes) -> char* {
    char* r = p;
    p += (bytes + 255) & ~(size_t)255;
    return r;
  };
  _Float16* Wq_t  = (_Float16*)alloc((size_t)L_ * E_ * E_ * 2);
  _Float16* Wk_t  = (_Float16*)alloc((size_t)L_ * E_ * E_ * 2);
  _Float16* Wv_t  = (_Float16*)alloc((size_t)L_ * E_ * E_ * 2);
  _Float16* Wo_t  = (_Float16*)alloc((size_t)L_ * E_ * E_ * 2);
  _Float16* W1_t  = (_Float16*)alloc((size_t)L_ * E_ * FF_ * 2);
  _Float16* W2_t  = (_Float16*)alloc((size_t)L_ * E_ * FF_ * 2);
  _Float16* Wlm_t = (_Float16*)alloc((size_t)VPAD * E_ * 2);
  float*    x     = (float*)alloc((size_t)MTOK * E_ * 4);
  _Float16* h16   = (_Float16*)alloc((size_t)MTOK * E_ * 2);
  _Float16* q16   = (_Float16*)alloc((size_t)MTOK * E_ * 2);
  _Float16* k16   = (_Float16*)alloc((size_t)MTOK * E_ * 2);
  _Float16* v16   = (_Float16*)alloc((size_t)MTOK * E_ * 2);
  _Float16* a16   = (_Float16*)alloc((size_t)MTOK * E_ * 2);
  _Float16* ff16  = (_Float16*)alloc((size_t)MTOK * FF_ * 2);

  // ---- weight convert + transpose (f32 [K,N] -> f16 [Npad][K]) ----
  const int gEE = (E_ * E_ + 255) / 256;
  const int gEF = (E_ * FF_ + 255) / 256;
  for (int l = 0; l < L_; ++l) {
    size_t oEE = (size_t)l * E_ * E_;
    size_t oEF = (size_t)l * E_ * FF_;
    k_convT<<<gEE, 256, 0, stream>>>(Wq + oEE, Wq_t + oEE, E_, E_, E_);
    k_convT<<<gEE, 256, 0, stream>>>(Wk + oEE, Wk_t + oEE, E_, E_, E_);
    k_convT<<<gEE, 256, 0, stream>>>(Wv + oEE, Wv_t + oEE, E_, E_, E_);
    k_convT<<<gEE, 256, 0, stream>>>(Wo + oEE, Wo_t + oEE, E_, E_, E_);
    k_convT<<<gEF, 256, 0, stream>>>(W1 + oEF, W1_t + oEF, E_, FF_, FF_);
    k_convT<<<gEF, 256, 0, stream>>>(W2 + oEF, W2_t + oEF, FF_, E_, E_);
  }
  k_convT<<<(VPAD * E_ + 255) / 256, 256, 0, stream>>>(Wlm, Wlm_t, E_, V_, VPAD);

  // ---- embedding ----
  k_embed<<<MTOK, 128, 0, stream>>>(tok, pos, idx, x);

  const dim3 gQKV(MTOK / 64, E_ / 128);   // 512 x 3
  const dim3 gFF(MTOK / 64, FF_ / 128);   // 512 x 12
  const dim3 gLM(MTOK / 64, VPAD / 128);  // 512 x 1
  const dim3 gAtt(TC_ / 16, H_, B_);      // 16 x 6 x 128

  for (int l = 0; l < L_; ++l) {
    size_t oEE = (size_t)l * E_ * E_;
    size_t oEF = (size_t)l * E_ * FF_;
    // LN1 -> h16
    k_ln<<<MTOK, 128, 0, stream>>>(x, ln1s + l * E_, ln1b + l * E_, h16);
    // Q, K, V projections (f16 outputs)
    k_gemm<<<gQKV, 128, 0, stream>>>(h16, Wq_t + oEE, nullptr, nullptr,
                                     nullptr, q16, E_, E_, 0);
    k_gemm<<<gQKV, 128, 0, stream>>>(h16, Wk_t + oEE, nullptr, nullptr,
                                     nullptr, k16, E_, E_, 0);
    k_gemm<<<gQKV, 128, 0, stream>>>(h16, Wv_t + oEE, nullptr, nullptr,
                                     nullptr, v16, E_, E_, 0);
    // causal attention
    k_attn<<<gAtt, 32, 0, stream>>>(q16, k16, v16, a16);
    // x += attn @ Wo + bo
    k_gemm<<<gQKV, 128, 0, stream>>>(a16, Wo_t + oEE, bo + l * E_, x, x,
                                     nullptr, E_, E_, 0);
    // LN2 -> h16
    k_ln<<<MTOK, 128, 0, stream>>>(x, ln2s + l * E_, ln2b + l * E_, h16);
    // ff = relu(h @ W1 + b1)  (f16)
    k_gemm<<<gFF, 128, 0, stream>>>(h16, W1_t + oEF, b1 + l * FF_, nullptr,
                                    nullptr, ff16, E_, FF_, 1);
    // x += ff @ W2 + b2
    k_gemm<<<gQKV, 128, 0, stream>>>(ff16, W2_t + oEF, b2 + l * E_, x, x,
                                     nullptr, FF_, E_, 0);
  }

  // final LN + LM head -> logits (f32 into d_out)
  k_ln<<<MTOK, 128, 0, stream>>>(x, lnfs, lnfb, h16);
  k_gemm<<<gLM, 128, 0, stream>>>(h16, Wlm_t, blm, nullptr, logits, nullptr,
                                  E_, V_, 0);

  // loss
  k_zero<<<1, 32, 0, stream>>>(loss);
  k_loss<<<(MTOK + 255) / 256, 256, 0, stream>>>(logits, tgt, loss);
}
